// SoftAggBasic_37692632990244
// MI455X (gfx1250) — compile-verified, matrix operand order checked by reading, not run on hardware
//
#include <hip/hip_runtime.h>
#include <hip/hip_bf16.h>

typedef __attribute__((ext_vector_type(16))) _Float16 v16h;
typedef __attribute__((ext_vector_type(8)))  float    v8f;

#define B_  8
#define N_  65536
#define D_  512
#define S_  1024
#define BN_ (B_ * N_)        // 524288
#define BS_ (B_ * S_)        // 8192
#define BSD_ ((size_t)BS_ * D_)  // 4,194,304 floats (16 MB)

// ---------------- float atomic max (sign-split trick) ----------------
__device__ __forceinline__ void atomicMaxF32(float* addr, float val) {
  if (val >= 0.0f) {
    atomicMax((int*)addr, __float_as_int(val));
  } else {
    atomicMin((unsigned int*)addr, __float_as_uint(val));
  }
}

// ---------------- K0: init m=-inf, denom=0, xbar=0 ----------------
__global__ void k0_init(float* __restrict__ m, float* __restrict__ denom,
                        float* __restrict__ xbar) {
  size_t i = (size_t)blockIdx.x * blockDim.x + threadIdx.x;
  if (i < BSD_) xbar[i] = 0.0f;
  if (i < BS_) { denom[i] = 0.0f; m[i] = -__builtin_inff(); }
}

// ---------------- K1: logits = x.Wg + bg ; seg-max -> m ----------------
// one wave (32 lanes) per row of x [B*N, D]
__global__ void k1_logits(const float* __restrict__ x, const float* __restrict__ Wg,
                          const float* __restrict__ bg, const int* __restrict__ ix,
                          float* __restrict__ logits, float* __restrict__ m) {
  int wid  = threadIdx.x >> 5;
  int lane = threadIdx.x & 31;
  int r = blockIdx.x * 8 + wid;            // row in [0, B*N)
  const float4* xp = (const float4*)(x + (size_t)r * D_);
  const float4* gp = (const float4*)Wg;
  float s = 0.0f;
#pragma unroll
  for (int j = 0; j < 4; ++j) {
    float4 xv = xp[lane + 32 * j];
    float4 gv = gp[lane + 32 * j];
    s += xv.x * gv.x + xv.y * gv.y + xv.z * gv.z + xv.w * gv.w;
  }
#pragma unroll
  for (int o = 16; o > 0; o >>= 1) s += __shfl_xor(s, o, 32);
  if (lane == 0) {
    float lg = s + bg[0];
    logits[r] = lg;
    int b = r >> 16;            // r / N_
    int n = r & (N_ - 1);
    int seg = ix[n];
    atomicMaxF32(&m[b * S_ + seg], lg);
  }
}

// ---------------- K2: e = exp(logits - m); denom += e ----------------
__global__ void k2_exp(float* __restrict__ logits, const int* __restrict__ ix,
                       const float* __restrict__ m, float* __restrict__ denom) {
  int r = blockIdx.x * blockDim.x + threadIdx.x;   // [0, B*N)
  int b = r >> 16;
  int n = r & (N_ - 1);
  int seg = ix[n];
  float ev = __expf(logits[r] - m[b * S_ + seg]);
  logits[r] = ev;                                   // store e in place
  atomicAdd(&denom[b * S_ + seg], ev);
}

// ---------------- K3: xbar[b,seg,:] += (e/denom) * x[b,n,:] ----------------
// one wave per row; f32 atomics land in the 16MB L2-resident xbar
__global__ void k3_accum(const float* __restrict__ x, const float* __restrict__ e,
                         const int* __restrict__ ix, const float* __restrict__ denom,
                         float* __restrict__ xbar) {
  int wid  = threadIdx.x >> 5;
  int lane = threadIdx.x & 31;
  int r = blockIdx.x * 8 + wid;
  int b = r >> 16;
  int n = r & (N_ - 1);
  int seg = ix[n];
  float w = e[r] / denom[b * S_ + seg];
  const float4* xp = (const float4*)(x + (size_t)r * D_);
  float* xb = xbar + ((size_t)(b * S_ + seg)) * D_;
#pragma unroll
  for (int j = 0; j < 4; ++j) {
    int idx = lane + 32 * j;
    float4 xv = xp[idx];
    atomicAdd(xb + 4 * idx + 0, w * xv.x);
    atomicAdd(xb + 4 * idx + 1, w * xv.y);
    atomicAdd(xb + 4 * idx + 2, w * xv.z);
    atomicAdd(xb + 4 * idx + 3, w * xv.w);
  }
}

// ---------------- K4/K5: Out[M,512] = A[M,512] @ W[512,512] + bias ----------------
// one wave per 16x16 output tile, v_wmma_f32_16x16x32_f16 over K in steps of 32.
// Unroll-4 lets the scheduler pipeline global loads + v_cvt_pk_f16_f32 against
// four in-flight WMMAs (XDL co-execution), hiding L2 latency on the B-column loads.
__global__ void gemm_wmma(const float* __restrict__ A, const float* __restrict__ W,
                          const float* __restrict__ bias, float* __restrict__ Out) {
  int wid  = threadIdx.x >> 5;
  int lane = threadIdx.x & 31;
  int gw = blockIdx.x * 8 + wid;    // 16384 waves total for M=8192
  int tm = gw >> 5;                 // row tile  [0,512)
  int tn = gw & 31;                 // col tile  [0,32)
  int rc   = lane & 15;
  int half = lane >> 4;
  int mrow = tm * 16 + rc;          // A row held by this lane
  int ncol = tn * 16 + rc;          // B/D column held by this lane

  v8f acc = {};
#pragma unroll 4
  for (int kk = 0; kk < D_; kk += 32) {
    int base = kk + half * 8;       // this lane's K block: base..base+7, base+16..base+23
    const float4* ap  = (const float4*)(A + (size_t)mrow * D_ + base);
    float4 a0 = ap[0], a1 = ap[1];
    const float4* ap2 = (const float4*)(A + (size_t)mrow * D_ + base + 16);
    float4 a2 = ap2[0], a3 = ap2[1];

    v16h av, bv;
    av[0]=(_Float16)a0.x; av[1]=(_Float16)a0.y; av[2]=(_Float16)a0.z; av[3]=(_Float16)a0.w;
    av[4]=(_Float16)a1.x; av[5]=(_Float16)a1.y; av[6]=(_Float16)a1.z; av[7]=(_Float16)a1.w;
    av[8]=(_Float16)a2.x; av[9]=(_Float16)a2.y; av[10]=(_Float16)a2.z; av[11]=(_Float16)a2.w;
    av[12]=(_Float16)a3.x; av[13]=(_Float16)a3.y; av[14]=(_Float16)a3.z; av[15]=(_Float16)a3.w;

    const float* wp = W + ncol;     // column ncol, rows k (stride 512) — 1MB matrix, L2-hot
#pragma unroll
    for (int e = 0; e < 16; ++e) {
      int k = base + (e < 8 ? e : e + 8);
      bv[e] = (_Float16)wp[(size_t)k * D_];
    }
    acc = __builtin_amdgcn_wmma_f32_16x16x32_f16(
        false, av, false, bv, (short)0, acc, false, false);
  }

  float bb = bias[ncol];
#pragma unroll
  for (int r8 = 0; r8 < 8; ++r8) {
    int m = tm * 16 + half * 8 + r8;
    Out[(size_t)m * D_ + ncol] = acc[r8] + bb;
  }
}

// ---------------- K6: out[b,n,:] = t[b, ix[n], :] (float4 gather) ----------------
__global__ void k6_gather(const float* __restrict__ t, const int* __restrict__ ix,
                          float* __restrict__ out) {
  size_t i = (size_t)blockIdx.x * blockDim.x + threadIdx.x;  // over B*N*D/4
  size_t row = i >> 7;               // D/4 = 128 float4 per row
  int    c   = (int)(i & 127);
  int    b   = (int)(row >> 16);     // row / N_
  int    n   = (int)(row & (N_ - 1));
  int    seg = ix[n];
  const float4* src = (const float4*)(t + ((size_t)(b * S_ + seg)) * D_);
  ((float4*)out)[i] = src[c];
}

extern "C" void kernel_launch(void* const* d_in, const int* in_sizes, int n_in,
                              void* d_out, int out_size, void* d_ws, size_t ws_size,
                              hipStream_t stream) {
  const float* x  = (const float*)d_in[0];
  const int*   ix = (const int*)  d_in[1];
  const float* Wf = (const float*)d_in[2];
  const float* bf = (const float*)d_in[3];
  const float* Wg = (const float*)d_in[4];
  const float* bg = (const float*)d_in[5];
  const float* Wh = (const float*)d_in[6];
  const float* bh = (const float*)d_in[7];
  float* out = (float*)d_out;

  float* ws     = (float*)d_ws;
  float* logits = ws;                        // B*N      (reused as e)
  float* m      = logits + BN_;              // B*S
  float* denom  = m + BS_;                   // B*S
  float* xbar   = denom + BS_;               // B*S*D    (reused as t)
  float* z      = xbar + BSD_;               // B*S*D

  // K0: init (covers xbar zeroing + m/denom init)
  k0_init<<<(unsigned)((BSD_ + 255) / 256), 256, 0, stream>>>(m, denom, xbar);
  // K1: logits + segment max (wave per row, 8 waves/block)
  k1_logits<<<BN_ / 8, 256, 0, stream>>>(x, Wg, bg, ix, logits, m);
  // K2: exp + denom
  k2_exp<<<BN_ / 256, 256, 0, stream>>>(logits, ix, m, denom);
  // K3: weighted scatter-accumulate into xbar
  k3_accum<<<BN_ / 8, 256, 0, stream>>>(x, logits, ix, denom, xbar);
  // K4: z = xbar @ Wf + bf   (WMMA)
  gemm_wmma<<<(BS_ / 16) * (D_ / 16) / 8, 256, 0, stream>>>(xbar, Wf, bf, z);
  // K5: t = z @ Wh + bh      (WMMA; t reuses xbar storage)
  gemm_wmma<<<(BS_ / 16) * (D_ / 16) / 8, 256, 0, stream>>>(z, Wh, bh, xbar);
  // K6: gather rows back to [B, N, D]
  k6_gather<<<(unsigned)(((size_t)BN_ * D_ / 4) / 256), 256, 0, stream>>>(xbar, ix, out);
}
